// NeuronMemory_21157008900529
// MI455X (gfx1250) — matrix operand order repaired and checked by reading, not compiled, not run on hardware
//
#include <hip/hip_runtime.h>

#define B_    4
#define S_    2048
#define D_    1024
#define NC_   16
#define NK_   32768
#define R_    128
#define TOPK_ 8

typedef __attribute__((ext_vector_type(16))) __bf16 vbf16;
typedef __attribute__((ext_vector_type(8)))  float  v8f;
typedef __attribute__((ext_vector_type(4)))  unsigned int v4u;
typedef __attribute__((ext_vector_type(8)))  int  v8i;
typedef __attribute__((ext_vector_type(4)))  int  v4i;

struct U2 { unsigned int x, y; };
struct U4 { unsigned int x, y, z, w; };

union V16 {
  vbf16 v;
  U4    q[2];
  unsigned int u[8];
};

__device__ __forceinline__ unsigned short f2bf(float f) {
  unsigned int u = __float_as_uint(f);
  u += 0x7FFFu + ((u >> 16) & 1u);          // round-to-nearest-even
  return (unsigned short)(u >> 16);
}
__device__ __forceinline__ unsigned int pk2bf(float lo, float hi) {
  return (unsigned int)f2bf(lo) | ((unsigned int)f2bf(hi) << 16);
}
// wave32 half-swap: exchange with lane^16 (group-of-32 swizzle, xor=0x10,and=0x1f)
__device__ __forceinline__ int   swz16i(int v)   { return __builtin_amdgcn_ds_swizzle(v, 0x401F); }
__device__ __forceinline__ float swz16f(float v) { return __int_as_float(__builtin_amdgcn_ds_swizzle(__float_as_int(v), 0x401F)); }

// ---------------------------------------------------------------------------
// TDM: load a 128x128 bf16 tile (row stride 128 elems) into LDS with a 16B pad
// after every 256B row  ->  LDS row stride 272B.  D# per CDNA5 ISA §8.3/8.4.
// Issued by one wave; EXEC is ignored by TENSOR ops; tracked by TENSORcnt.
// 6-arg builtin form (clang-23 / therock): (g0, g1, g2, g3, g_extra, cpol).
// ---------------------------------------------------------------------------
__device__ __forceinline__ void tdm_load_k_tile(const unsigned short* gptr, unsigned ldsoff) {
  unsigned long long ga = (unsigned long long)gptr;
  v4u g0;
  g0[0] = 1u;                                              // count=1, load, user D#
  g0[1] = ldsoff;                                          // lds_addr (bytes)
  g0[2] = (unsigned)ga;                                    // global_addr[31:0]
  g0[3] = (unsigned)((ga >> 32) & 0x01FFFFFFu) | (2u << 30); // addr[56:32] | type=2
  v8i g1 = {0, 0, 0, 0, 0, 0, 0, 0};
  // data_size=2B (code 1) | pad_enable | pad_interval=64dw (code 5) | pad_amount=4dw (code 3)
  g1[0] = (int)((1u << 16) | (1u << 20) | (5u << 22) | (3u << 25));
  g1[1] = (int)(128u << 16);        // tensor_dim0 = 128  (bits [79:48], low16)
  g1[2] = (int)0x80000000u;         // tensor_dim1 = 32768 (bits [111:80], low16)
  g1[3] = (int)(128u << 16);        // tile_dim0 = 128    (bits [127:112])
  g1[4] = 128;                      // tile_dim1 = 128 rows (bits [143:128])
  g1[5] = 128;                      // tensor_dim0_stride = 128 elems (bits [207:160])
  v4i z4 = {0, 0, 0, 0};
  v8i z8 = {0, 0, 0, 0, 0, 0, 0, 0};
  __builtin_amdgcn_tensor_load_to_lds(g0, g1, z4, z4, z8, 0);
}

// ---------------------------------------------------------------------------
// Kernel 1: SC_T[b,r,d] = sum_n mw[b,n]*cn[n,d,r]  -> bf16, stored TRANSPOSED
// so the Q-GEMM B-operand (K-pairs at fixed r) is contiguous.
// ---------------------------------------------------------------------------
__global__ void __launch_bounds__(256) sc_kernel(const float* __restrict__ mw,
                                                 const float* __restrict__ cn,
                                                 unsigned short* __restrict__ scT) {
  int gid = blockIdx.x * 256 + threadIdx.x;   // B*D*R = 524288
  int b  = gid >> 17;                         // D*R = 131072
  int dr = gid & (131072 - 1);
  int d  = dr >> 7, r = dr & 127;
  float acc = 0.f;
#pragma unroll
  for (int n = 0; n < NC_; ++n)
    acc += mw[b * NC_ + n] * cn[n * (D_ * R_) + dr];
  scT[b * (D_ * R_) + r * D_ + d] = f2bf(acc);
}

// ---------------------------------------------------------------------------
// Kernel 1b: knowledge_K f32 -> bf16 (one-time; scores kernel then streams
// bf16 tiles straight to LDS via TDM, no per-stage conversion).
// ---------------------------------------------------------------------------
__global__ void __launch_bounds__(256) kconv_kernel(const float* __restrict__ kf,
                                                    unsigned short* __restrict__ kbf) {
  int gid = blockIdx.x * 256 + threadIdx.x;   // NK*R/4 float4 = 1M
  float4 f = ((const float4*)kf)[gid];
  U2 p; p.x = pk2bf(f.x, f.y); p.y = pk2bf(f.z, f.w);
  ((U2*)kbf)[gid] = p;
}

// ---------------------------------------------------------------------------
// Kernel 2: Q[b,s,r] = (x @ SC) * (1/sqrt(R))  via WMMA bf16, stored bf16
// Block = 16 tokens (staged in LDS as bf16), 8 waves cover the 8 r-tiles.
// ---------------------------------------------------------------------------
__global__ void __launch_bounds__(256) q_kernel(const float* __restrict__ x,
                                                const unsigned short* __restrict__ scT,
                                                unsigned short* __restrict__ qbf) {
  __shared__ __align__(16) unsigned char lds[16 * 2064];   // 16 rows x 1024 bf16, +16B pad
  const int b   = blockIdx.x >> 7;
  const int t0  = (blockIdx.x & 127) << 4;
  const int tid = threadIdx.x;

  const float* xb = x + (size_t)(b * S_ + t0) * D_;
#pragma unroll
  for (int j = 0; j < 16; ++j) {              // 4096 float4 over 256 threads
    int l   = j * 256 + tid;
    int row = l >> 8;                         // 256 float4 per row
    int c4  = l & 255;
    float4 f = ((const float4*)(xb + row * D_))[c4];
    U2 p; p.x = pk2bf(f.x, f.y); p.y = pk2bf(f.z, f.w);
    *(U2*)(lds + row * 2064 + c4 * 8) = p;
  }
  __syncthreads();

  const int wave = tid >> 5, lane = tid & 31, h = lane >> 4, m16 = lane & 15;
  const int r0 = wave << 4;
  const unsigned short* scr = scT + (size_t)b * (D_ * R_) + (size_t)(r0 + m16) * D_;
  v8f acc = {};
#pragma unroll 4
  for (int kc = 0; kc < 32; ++kc) {           // K = D in chunks of 32
    const int k0 = kc * 32;
    V16 a, bm;
    const unsigned char* arow = lds + m16 * 2064;
    a.q[0] = *(const U4*)(arow + (k0 + 16 * h) * 2);       // K = k0+16h .. +7
    a.q[1] = *(const U4*)(arow + (k0 + 16 * h + 8) * 2);   // K = k0+16h+8 .. +7
    const U4* bp = (const U4*)(scr + k0 + 16 * h);         // K = k0+16h .. +15 contiguous
    bm.q[0] = bp[0]; bm.q[1] = bp[1];
    acc = __builtin_amdgcn_wmma_f32_16x16x32_bf16(false, a.v, false, bm.v,
                                                  (short)0, acc, false, false);
  }
  const float qscale = 0.088388347648318447f;  // 1/sqrt(128): fold score scale into Q
#pragma unroll
  for (int j = 0; j < 8; ++j) {               // D-layout: M = j + 8h, N = m16
    int m = t0 + j + 8 * h;
    qbf[(size_t)(b * S_ + m) * R_ + r0 + m16] = f2bf(acc[j] * qscale);
  }
}

// ---------------------------------------------------------------------------
// Kernel 3 (fused core): scores = Q @ K^T via WMMA bf16 (M=knowledge, N=token),
// TDM double-buffered K streaming, running per-lane top-8, half-wave bitonic
// merge, softmax, write idx+weights.  Block = 128 tokens (8 waves x 16).
// ---------------------------------------------------------------------------
#define TILE_BYTES (128 * 272)

__global__ void __launch_bounds__(256) scores_kernel(const unsigned short* __restrict__ qbf,
                                                     const unsigned short* __restrict__ kbf,
                                                     int* __restrict__ out_idx,
                                                     float* __restrict__ out_w) {
  __shared__ __align__(16) unsigned char lds[2][TILE_BYTES];
  const int tid  = threadIdx.x;
  const int b    = blockIdx.x >> 4;
  const int tb   = (blockIdx.x & 15) << 7;
  const int wave = tid >> 5, lane = tid & 31, h = lane >> 4, m16 = lane & 15;
  const int tok  = tb + wave * 16 + m16;      // this lane's token column

  // Resident Q (B operand): 4 chunks, contiguous 32B per chunk per lane
  V16 qc[4];
  {
    const unsigned short* qrow = qbf + (size_t)(b * S_ + tok) * R_;
#pragma unroll
    for (int c = 0; c < 4; ++c) {
      const U4* p = (const U4*)(qrow + c * 32 + 16 * h);
      qc[c].q[0] = p[0]; qc[c].q[1] = p[1];
    }
  }

  float key[8]; int kid[8];
#pragma unroll
  for (int j = 0; j < 8; ++j) { key[j] = -3.0e38f; kid[j] = 0; }

  // LDS byte offset of the staging buffers (flat LDS-aperture low 32 bits).
  const unsigned ldsbase = (unsigned)(size_t)(const void*)&lds[0][0];

  if (tid < 32) tdm_load_k_tile(kbf, ldsbase);          // prologue: buf0 <- rows [0,128)

  int buf = 0;
  for (int kb = 0; kb < NK_; kb += 128, buf ^= 1) {
    if (tid < 32) __builtin_amdgcn_s_wait_tensorcnt((short)0);
    __syncthreads();                                    // current tile resident
    if ((tid < 32) && (kb + 128 < NK_))                 // overlap next DMA with compute
      tdm_load_k_tile(kbf + (size_t)(kb + 128) * R_, ldsbase + (unsigned)((buf ^ 1) * TILE_BYTES));

    const unsigned char* tbase = &lds[buf][0];
#pragma unroll 1
    for (int st = 0; st < 8; ++st) {          // 8 sub-tiles of 16 knowledge rows
      const unsigned char* arow = tbase + (st * 16 + m16) * 272;
      v8f acc = {};
#pragma unroll
      for (int c = 0; c < 4; ++c) {
        V16 a;
        a.q[0] = *(const U4*)(arow + (c * 32 + 16 * h) * 2);
        a.q[1] = *(const U4*)(arow + (c * 32 + 16 * h + 8) * 2);
        acc = __builtin_amdgcn_wmma_f32_16x16x32_bf16(false, a.v, false, qc[c].v,
                                                      (short)0, acc, false, false);
      }
      const int mbase = kb + st * 16 + 8 * h; // this lane's knowledge-row base
#pragma unroll
      for (int j = 0; j < 8; ++j) {
        float s = acc[j];
        if (s > key[7]) {                     // rare: branch usually skipped
          int id = mbase + j;
#pragma unroll
          for (int p = 7; p >= 1; --p) {
            bool  up = s > key[p - 1];
            float nk = up ? key[p - 1] : s;
            int   ni = up ? kid[p - 1] : id;
            bool  cp = s > key[p];
            key[p] = cp ? nk : key[p];
            kid[p] = cp ? ni : kid[p];
          }
          if (s > key[0]) { key[0] = s; kid[0] = id; }
        }
      }
    }
    __syncthreads();                                    // done reading lds[buf]
  }

  // Merge the two half-wave lists (disjoint knowledge subsets, same token):
  // concat desc-list with reversed partner list -> bitonic, 4-stage merge.
  float mk[16]; int mi[16];
#pragma unroll
  for (int j = 0; j < 8; ++j) { mk[j] = key[j]; mi[j] = kid[j]; }
#pragma unroll
  for (int j = 0; j < 8; ++j) {
    mk[8 + j] = swz16f(key[7 - j]);
    mi[8 + j] = swz16i(kid[7 - j]);
  }
#pragma unroll
  for (int d = 8; d >= 1; d >>= 1) {
#pragma unroll
    for (int i = 0; i < 16; ++i) {
      if (!(i & d)) {
        bool  sw = mk[i] < mk[i + d];
        float tk = sw ? mk[i + d] : mk[i];
        mk[i + d] = sw ? mk[i] : mk[i + d];
        mk[i] = tk;
        int ti = sw ? mi[i + d] : mi[i];
        mi[i + d] = sw ? mi[i] : mi[i + d];
        mi[i] = ti;
      }
    }
  }

  if (h == 0) {                               // lanes 0..15 own the final lists
    float mx = mk[0], e[8], sum = 0.f;
#pragma unroll
    for (int j = 0; j < 8; ++j) { e[j] = __expf(mk[j] - mx); sum += e[j]; }
    float inv = 1.0f / sum;
    int base = (b * S_ + tok) * TOPK_;
#pragma unroll
    for (int j = 0; j < 8; ++j) {
      out_idx[base + j] = mi[j];
      out_w[base + j]   = e[j] * inv;
    }
  }
}

// ---------------------------------------------------------------------------
// Kernel 4: out[b,s,:] = sum_k w_k * V[idx_k,:]   (one block per token)
// ---------------------------------------------------------------------------
__global__ void __launch_bounds__(256) gather_kernel(const float* __restrict__ V,
                                                     const int* __restrict__ idx,
                                                     const float* __restrict__ w,
                                                     float* __restrict__ out) {
  const int token = blockIdx.x;
  const int tid   = threadIdx.x;
  const int base  = token * TOPK_;
#pragma unroll
  for (int k = 0; k < TOPK_; ++k)
    __builtin_prefetch(V + (size_t)idx[base + k] * D_ + tid * 4, 0, 1);
  float4 acc = make_float4(0.f, 0.f, 0.f, 0.f);
#pragma unroll
  for (int k = 0; k < TOPK_; ++k) {
    const float4* vr = (const float4*)(V + (size_t)idx[base + k] * D_);
    float4 v = vr[tid];
    float wk = w[base + k];
    acc.x += wk * v.x; acc.y += wk * v.y; acc.z += wk * v.z; acc.w += wk * v.w;
  }
  ((float4*)(out + (size_t)token * D_))[tid] = acc;
}

// ---------------------------------------------------------------------------
extern "C" void kernel_launch(void* const* d_in, const int* in_sizes, int n_in,
                              void* d_out, int out_size, void* d_ws, size_t ws_size,
                              hipStream_t stream) {
  (void)in_sizes; (void)n_in; (void)out_size; (void)ws_size;
  const float* x  = (const float*)d_in[0];
  const float* mw = (const float*)d_in[1];
  const float* cn = (const float*)d_in[2];
  const float* kK = (const float*)d_in[3];
  const float* kV = (const float*)d_in[4];

  unsigned short* scT = (unsigned short*)d_ws;                                   // 1 MiB
  unsigned short* qbf = (unsigned short*)((char*)d_ws + (size_t)(1 << 20));      // 2 MiB
  unsigned short* kbf = (unsigned short*)((char*)d_ws + (size_t)(3 << 20));      // 8 MiB

  float* outp    = (float*)d_out;                                 // [B,S,D]
  int*   out_idx = (int*)  (outp + (size_t)B_*S_*D_);             // [B,S,8] int32
  float* out_w   = (float*)(outp + (size_t)B_*S_*D_ + (size_t)B_*S_*TOPK_); // [B,S,8]

  hipLaunchKernelGGL(sc_kernel,     dim3((B_*D_*R_)/256),  dim3(256), 0, stream, mw, cn, scT);
  hipLaunchKernelGGL(kconv_kernel,  dim3((NK_*R_)/1024),   dim3(256), 0, stream, kK, kbf);
  hipLaunchKernelGGL(q_kernel,      dim3(B_*(S_/16)),      dim3(256), 0, stream, x, scT, qbf);
  hipLaunchKernelGGL(scores_kernel, dim3((B_*S_)/128),     dim3(256), 0, stream, qbf, kbf, out_idx, out_w);
  hipLaunchKernelGGL(gather_kernel, dim3(B_*S_),           dim3(256), 0, stream, kV, out_idx, out_w, outp);
}
